// TranceptionAttention_19902878450197
// MI455X (gfx1250) — matrix operand order repaired and checked by reading, not compiled, not run on hardware
//
#include <hip/hip_runtime.h>
#include <stdint.h>

// B=2, L=2048, E=1024, H=16, HD=64, G=4. M = B*L = 4096.
#define LSEQ  2048
#define EDIM  1024
#define MROWS 4096

typedef __attribute__((ext_vector_type(16))) __bf16 v16bf;
typedef __attribute__((ext_vector_type(8)))  __bf16 v8bf;
typedef __attribute__((ext_vector_type(8)))  float  v8f;

// ---------------------------------------------------------------------------
// Fragment loaders (bf16, K-major rows, 16B-aligned)
// A 16x32 layout (ISA 7.12.2): lanes 0-15 hold K[0..7],K[16..23]; lanes 16-31
// hold K[8..15],K[24..31] -> per lane: two 8-element chunks at +half*8, +16+half*8.
// B 32x16 layout (same K order as A): lanes 0-15 hold K[0..7],K[16..23];
// lanes 16-31 hold K[8..15],K[24..31] -> per lane: two 8-element chunks at +half*8, +16+half*8.
// ---------------------------------------------------------------------------
__device__ __forceinline__ v16bf cat8(v8bf a, v8bf b) {
  return __builtin_shufflevector(a, b, 0,1,2,3,4,5,6,7,8,9,10,11,12,13,14,15);
}
__device__ __forceinline__ v16bf load_a_frag(const __bf16* p) {
  v8bf c0 = *(const v8bf*)(p);
  v8bf c1 = *(const v8bf*)(p + 16);
  return cat8(c0, c1);
}
__device__ __forceinline__ v16bf load_b_frag(const __bf16* p) {
  v8bf c0 = *(const v8bf*)(p);
  v8bf c1 = *(const v8bf*)(p + 16);
  return cat8(c0, c1);
}

// ---------------------------------------------------------------------------
// bf16x3 WMMA GEMM: C[m,n] = sum_k A[m,k]*B[n,k] (+bias[n])
// A,B given as bf16 hi/lo splits of fp32 (a ~= hi + lo). Product approximated
// by hi*hi + hi*lo + lo*hi (error ~2^-16 relative).
// Wave tile 64x64 (4x4 WMMA 16x16 accumulators). Block = 8 waves covering
// 64(M) x 512(N). Grid = (N/512, M/64). Dims must divide evenly (they do).
// ---------------------------------------------------------------------------
__global__ __launch_bounds__(256) void gemm_bf16x3(
    const __bf16* __restrict__ Ahi, const __bf16* __restrict__ Alo,
    const __bf16* __restrict__ Bhi, const __bf16* __restrict__ Blo,
    const float* __restrict__ bias, float* __restrict__ C,
    int M, int N, int K)
{
  const int lane = threadIdx.x & 31;
  const int wave = threadIdx.x >> 5;
  const int half = lane >> 4;   // which 16-lane half
  const int l15  = lane & 15;
  const int m0 = blockIdx.y * 64;
  const int n0 = blockIdx.x * 512 + wave * 64;

  v8f zero;
#pragma unroll
  for (int j = 0; j < 8; ++j) zero[j] = 0.0f;

  v8f acc[4][4];
#pragma unroll
  for (int i = 0; i < 4; ++i)
#pragma unroll
    for (int j = 0; j < 4; ++j) acc[i][j] = zero;

  for (int k = 0; k < K; k += 32) {
    v16bf ah[4], al[4], bh[4], bl[4];
#pragma unroll
    for (int mt = 0; mt < 4; ++mt) {
      const size_t base = (size_t)(m0 + mt * 16 + l15) * K + k + half * 8;
      ah[mt] = load_a_frag(Ahi + base);
      al[mt] = load_a_frag(Alo + base);
      if (k + 32 < K) __builtin_prefetch(Ahi + base + 32, 0, 1);
    }
#pragma unroll
    for (int nt = 0; nt < 4; ++nt) {
      const size_t base = (size_t)(n0 + nt * 16 + l15) * K + k + half * 8;
      bh[nt] = load_b_frag(Bhi + base);
      bl[nt] = load_b_frag(Blo + base);
      if (k + 32 < K) __builtin_prefetch(Bhi + base + 32, 0, 1);
    }
#pragma unroll
    for (int mt = 0; mt < 4; ++mt)
#pragma unroll
      for (int nt = 0; nt < 4; ++nt) {
        acc[mt][nt] = __builtin_amdgcn_wmma_f32_16x16x32_bf16(
            false, ah[mt], false, bh[nt], (short)0, acc[mt][nt], false, false);
        acc[mt][nt] = __builtin_amdgcn_wmma_f32_16x16x32_bf16(
            false, ah[mt], false, bl[nt], (short)0, acc[mt][nt], false, false);
        acc[mt][nt] = __builtin_amdgcn_wmma_f32_16x16x32_bf16(
            false, al[mt], false, bh[nt], (short)0, acc[mt][nt], false, false);
      }
  }

  // Epilogue. C/D layout: VGPR j, lanes 0-15 -> M=j, lanes 16-31 -> M=8+j; N=lane&15.
#pragma unroll
  for (int nt = 0; nt < 4; ++nt) {
    const int col = n0 + nt * 16 + l15;
    const float bb = bias ? bias[col] : 0.0f;
#pragma unroll
    for (int mt = 0; mt < 4; ++mt) {
      const int rbase = m0 + mt * 16 + half * 8;
#pragma unroll
      for (int j = 0; j < 8; ++j)
        C[(size_t)(rbase + j) * N + col] = acc[mt][nt][j] + bb;
    }
  }
}

// ---------------------------------------------------------------------------
// fp32 -> bf16 hi/lo split helpers
// ---------------------------------------------------------------------------
__global__ void split_bf16(const float* __restrict__ src,
                           __bf16* __restrict__ hi, __bf16* __restrict__ lo, int n)
{
  int i = blockIdx.x * blockDim.x + threadIdx.x;
  const int stride = gridDim.x * blockDim.x;
  for (; i < n; i += stride) {
    float v = src[i];
    __bf16 h = (__bf16)v;
    hi[i] = h;
    lo[i] = (__bf16)(v - (float)h);
  }
}

// Mixed projection weights: rows 0..255 from Wv (branch-0 V path), rows
// 256..1023 from Wq (conv branches). Flat index select works since both are
// (1024,1024) row-major.
__global__ void build_wsel(const float* __restrict__ Wv, const float* __restrict__ Wq,
                           __bf16* __restrict__ hi, __bf16* __restrict__ lo, int n)
{
  int i = blockIdx.x * blockDim.x + threadIdx.x;
  const int stride = gridDim.x * blockDim.x;
  for (; i < n; i += stride) {
    float v = (i < 256 * 1024) ? Wv[i] : Wq[i];
    __bf16 h = (__bf16)v;
    hi[i] = h;
    lo[i] = (__bf16)(v - (float)h);
  }
}

// ---------------------------------------------------------------------------
// Combine the two chained depthwise convs into one causal conv of length
// 2k-1:  V[l] = cb + sum_t cw[t]*P[l-2(k-1)+t]  (zero-padded P),
// with boundary bias correction: for l < k-1 the intermediate y1 is
// zero-padded (not qb-padded), so subtract corr[(k-1)-l] = qb*sum_{j<(k-1)-l} vw[j].
// cw[t] = sum_{i+j=t} qw[i]*vw[j],  cb = vb + qb*sum_j vw[j].
// ---------------------------------------------------------------------------
__global__ void combine_conv(
    const float* q3w, const float* q3b, const float* v3w, const float* v3b,
    const float* q5w, const float* q5b, const float* v5w, const float* v5b,
    const float* q7w, const float* q7b, const float* v7w, const float* v7b,
    float* __restrict__ cw, float* __restrict__ cb, float* __restrict__ corr)
{
  const int t = threadIdx.x;
  if (t >= 192) return;
  const int br = t >> 6, d = t & 63;
  const float *qw, *qb, *vw, *vb;
  int ksz;
  if (br == 0)      { qw = q3w; qb = q3b; vw = v3w; vb = v3b; ksz = 3; }
  else if (br == 1) { qw = q5w; qb = q5b; vw = v5w; vb = v5b; ksz = 5; }
  else              { qw = q7w; qb = q7b; vw = v7w; vb = v7b; ksz = 7; }

  float qwl[7], vwl[7];
  for (int i = 0; i < 7; ++i) {
    qwl[i] = (i < ksz) ? qw[d * ksz + i] : 0.0f;
    vwl[i] = (i < ksz) ? vw[d * ksz + i] : 0.0f;
  }
  float S = 0.0f;
  for (int j = 0; j < 7; ++j) S += vwl[j];

  float* cwp = cw + (br * 64 + d) * 16;
  for (int tt = 0; tt < 16; ++tt) {
    float s = 0.0f;
    for (int i = 0; i < 7; ++i) {
      int j = tt - i;
      if (j >= 0 && j < 7) s += qwl[i] * vwl[j];
    }
    cwp[tt] = s;
  }
  cb[br * 64 + d] = vb[d] + qb[d] * S;

  float run = 0.0f;
  for (int p = 0; p < 8; ++p) {
    corr[(br * 64 + d) * 8 + p] = qb[d] * run;
    if (p < 7) run += vwl[p];
  }
}

// ---------------------------------------------------------------------------
// Fused conv + bf16 split epilogue producing GEMM2's A operand.
// Channels 0..255: identity copy of P. Channels 256..1023: fused causal conv.
// One block per (b,l) row; e strided by 256 -> branch uniform per wave.
// ---------------------------------------------------------------------------
__global__ __launch_bounds__(256) void dwconv_fused(
    const float* __restrict__ P, const float* __restrict__ cw,
    const float* __restrict__ cb, const float* __restrict__ corr,
    __bf16* __restrict__ Vhi, __bf16* __restrict__ Vlo)
{
  const int bl = blockIdx.x;            // b*L + l
  const int l  = bl & (LSEQ - 1);
  const int rowbase = bl << 10;         // *EDIM
  for (int e = threadIdx.x; e < EDIM; e += 256) {
    float v;
    if (e < 256) {
      v = P[rowbase + e];
    } else {
      const int br  = (e >> 8) - 1;     // 0,1,2
      const int d   = e & 63;
      const int ksz = 3 + 2 * br;
      const int off = 2 * (ksz - 1);
      const int taps = 2 * ksz - 1;
      const float* w = cw + (br * 64 + d) * 16;
      float acc = cb[br * 64 + d];
      if (l < ksz - 1) acc -= corr[(br * 64 + d) * 8 + (ksz - 1 - l)];
      int t0 = off - l; if (t0 < 0) t0 = 0;           // causal zero padding
      for (int tt = t0; tt < taps; ++tt)
        acc += w[tt] * P[rowbase + (tt - off) * EDIM + e];
      v = acc;
    }
    __bf16 h = (__bf16)v;
    Vhi[rowbase + e] = h;
    Vlo[rowbase + e] = (__bf16)(v - (float)h);
  }
}

// ---------------------------------------------------------------------------
extern "C" void kernel_launch(void* const* d_in, const int* in_sizes, int n_in,
                              void* d_out, int out_size, void* d_ws, size_t ws_size,
                              hipStream_t stream)
{
  (void)in_sizes; (void)n_in; (void)out_size; (void)ws_size;

  const float* x  = (const float*)d_in[0];
  const float* Wq = (const float*)d_in[1];
  // d_in[2] = Wk : unused (attention path multiplied by 0.0)
  const float* Wv = (const float*)d_in[3];
  const float* Wo = (const float*)d_in[4];
  const float* bo = (const float*)d_in[5];
  // conv params: 6:q3_w 7:q3_b 8:k3_w 9:k3_b 10:v3_w 11:v3_b, then k=5 at 12.., k=7 at 18..
  const float* q3w = (const float*)d_in[6];  const float* q3b = (const float*)d_in[7];
  const float* v3w = (const float*)d_in[10]; const float* v3b = (const float*)d_in[11];
  const float* q5w = (const float*)d_in[12]; const float* q5b = (const float*)d_in[13];
  const float* v5w = (const float*)d_in[16]; const float* v5b = (const float*)d_in[17];
  const float* q7w = (const float*)d_in[18]; const float* q7b = (const float*)d_in[19];
  const float* v7w = (const float*)d_in[22]; const float* v7b = (const float*)d_in[23];

  const int nX = MROWS * EDIM;     // 4,194,304
  const int nW = EDIM * EDIM;      // 1,048,576

  uintptr_t cur = (uintptr_t)d_ws;
  auto take = [&](size_t bytes) -> void* {
    uintptr_t r = cur;
    cur = (cur + bytes + 255) & ~(uintptr_t)255;
    return (void*)r;
  };

  __bf16* xhi  = (__bf16*)take((size_t)nX * 2);
  __bf16* xlo  = (__bf16*)take((size_t)nX * 2);
  __bf16* wshi = (__bf16*)take((size_t)nW * 2);
  __bf16* wslo = (__bf16*)take((size_t)nW * 2);
  __bf16* wohi = (__bf16*)take((size_t)nW * 2);
  __bf16* wolo = (__bf16*)take((size_t)nW * 2);
  float*  P    = (float*) take((size_t)nX * 4);
  __bf16* vhi  = (__bf16*)take((size_t)nX * 2);
  __bf16* vlo  = (__bf16*)take((size_t)nX * 2);
  float*  cw   = (float*) take(192 * 16 * 4);
  float*  cb   = (float*) take(192 * 4);
  float*  corr = (float*) take(192 * 8 * 4);

  // 1) precision splits + mixed projection weight
  split_bf16<<<2048, 256, 0, stream>>>(x, xhi, xlo, nX);
  build_wsel<<<1024, 256, 0, stream>>>(Wv, Wq, wshi, wslo, nW);
  split_bf16<<<1024, 256, 0, stream>>>(Wo, wohi, wolo, nW);
  combine_conv<<<1, 192, 0, stream>>>(q3w, q3b, v3w, v3b,
                                      q5w, q5b, v5w, v5b,
                                      q7w, q7b, v7w, v7b, cw, cb, corr);

  // 2) GEMM1: P = x @ Wsel.T   (4096 x 1024 x 1024)
  dim3 gg(EDIM / 512, MROWS / 64);
  gemm_bf16x3<<<gg, 256, 0, stream>>>(xhi, xlo, wshi, wslo, nullptr, P,
                                      MROWS, EDIM, EDIM);

  // 3) fused depthwise conv chain + bf16 split
  dwconv_fused<<<MROWS, 256, 0, stream>>>(P, cw, cb, corr, vhi, vlo);

  // 4) GEMM2: out = V @ Wo.T + bo
  gemm_bf16x3<<<gg, 256, 0, stream>>>(vhi, vlo, wohi, wolo, bo, (float*)d_out,
                                      MROWS, EDIM, EDIM);
}
